// ReinforcedDecoder_52175262712114
// MI455X (gfx1250) — compile-verified, matrix operand order checked by reading, not run on hardware
//
#include <hip/hip_runtime.h>
#include <hip/hip_bf16.h>

#define TGT  24
#define BSZ  16
#define SRCL 400
#define VOC  32000
#define DIM  256
#define FEAT 768   // 3*DIM
#define GDIM 1024  // 4*DIM gates
#define KXH  512   // concat [x|h] K

typedef __attribute__((ext_vector_type(16))) __bf16 v16bf;
typedef __attribute__((ext_vector_type(8)))  float  v8f;

union BfTile { uint4 q[2]; v16bf v; };

__device__ __forceinline__ unsigned short f2bf(float f) {
  unsigned u = __float_as_uint(f);
  u += 0x7FFFu + ((u >> 16) & 1u);          // round-to-nearest-even
  return (unsigned short)(u >> 16);
}

// Load 16 bf16 for one lane of a row-major 16xK A/B operand:
// two contiguous groups of 8 halves: [base..base+7] and [base+16..base+23].
__device__ __forceinline__ v16bf load_row16(const unsigned short* base) {
  BfTile t;
  t.q[0] = *(const uint4*)(base);
  t.q[1] = *(const uint4*)(base + 16);
  return t.v;
}

__device__ __forceinline__ float sigmoidf_fast(float x) {
  return 1.f / (1.f + __expf(-x));
}

// ---------------- one-time conversions / packings to bf16
// emb_bf   [VOC][DIM]      = bf16(W_emb)
// projT_bf [FEAT][DIM]     = bf16(W_proj^T)
// wg_bf    [GDIM][KXH]     = bf16([W_ih | W_hh]) row-major by gate j
// wencT_bf [DIM][DIM]      = bf16(W_attn_enc^T)  (row = output j)
// wdecT_bf [DIM][DIM]      = bf16(W_attn_dec^T)
__global__ void convert_kernel(const float* __restrict__ W_emb,
                               const float* __restrict__ W_proj,
                               const float* __restrict__ W_ih,
                               const float* __restrict__ W_hh,
                               const float* __restrict__ W_enc,
                               const float* __restrict__ W_dec,
                               unsigned short* __restrict__ emb_bf,
                               unsigned short* __restrict__ projT_bf,
                               unsigned short* __restrict__ wg_bf,
                               unsigned short* __restrict__ wencT_bf,
                               unsigned short* __restrict__ wdecT_bf) {
  int idx = blockIdx.x * blockDim.x + threadIdx.x;
  const int n0 = VOC * DIM;
  const int n1 = FEAT * DIM;
  const int n2 = GDIM * KXH;
  const int n3 = DIM * DIM;
  if (idx < n0) { emb_bf[idx] = f2bf(W_emb[idx]); return; }
  idx -= n0;
  if (idx < n1) {
    int row = idx / DIM, k = idx % DIM;      // projT[row][k] = W_proj[k][row]
    projT_bf[idx] = f2bf(W_proj[k * FEAT + row]);
    return;
  }
  idx -= n1;
  if (idx < n2) {
    int j = idx / KXH, k = idx % KXH;        // wg[j][0:256]=W_ih[j], [256:512]=W_hh[j]
    float v = (k < DIM) ? W_ih[j * DIM + k] : W_hh[j * DIM + (k - DIM)];
    wg_bf[idx] = f2bf(v);
    return;
  }
  idx -= n2;
  if (idx < n3) {
    int row = idx / DIM, k = idx % DIM;      // wencT[row][k] = W_enc[k][row]
    wencT_bf[idx] = f2bf(W_enc[k * DIM + row]);
    return;
  }
  idx -= n3;
  if (idx < n3) {
    int row = idx / DIM, k = idx % DIM;
    wdecT_bf[idx] = f2bf(W_dec[k * DIM + row]);
  }
}

// ---------------- one-time W_out = tanh(W_emb @ W_proj), stored bf16 (32000x768)
__global__ void wout_gemm_kernel(const unsigned short* __restrict__ emb_bf,
                                 const unsigned short* __restrict__ projT_bf,
                                 unsigned short* __restrict__ wout_bf) {
  int wave = threadIdx.x >> 5;
  int lane = threadIdx.x & 31;
  int id   = blockIdx.x * 4 + wave;          // 96000 tiles total
  int v0   = (id / (FEAT / 16)) * 16;
  int j0   = (id % (FEAT / 16)) * 16;
  int hi   = lane >> 4;
  int r16  = lane & 15;
  const unsigned short* ap = emb_bf   + (size_t)(v0 + r16) * DIM + hi * 8;
  const unsigned short* bp = projT_bf + (size_t)(j0 + r16) * DIM + hi * 8;
  v8f acc = {};
#pragma unroll
  for (int kb = 0; kb < DIM; kb += 32) {
    v16bf a = load_row16(ap + kb);
    v16bf b = load_row16(bp + kb);
    acc = __builtin_amdgcn_wmma_f32_16x16x32_bf16(false, a, false, b,
                                                  (short)0, acc, false, false);
  }
  int col = j0 + r16;
#pragma unroll
  for (int r = 0; r < 8; ++r) {
    int vrow = v0 + r + hi * 8;
    wout_bf[(size_t)vrow * FEAT + col] = f2bf(tanhf(acc[r]));
  }
}

// ---------------- state init
__global__ void init_kernel(const float* __restrict__ h0, const float* __restrict__ c0,
                            float* h, float* c, float* E_hist, float* hd_hist) {
  int idx = blockIdx.x * blockDim.x + threadIdx.x;
  if (idx < BSZ * DIM) { h[idx] = h0[idx]; c[idx] = c0[idx]; }
  if (idx < BSZ * SRCL) E_hist[idx] = 1.0f;
  if (idx < TGT * BSZ * DIM) hd_hist[idx] = 0.0f;
}

// ---------------- per-step serial part: WMMA LSTM + WMMA q/qd + attentions + feat
// 512 threads = 16 full wave32s; no EXEC divergence around any WMMA.
#define NT 512
__global__ void __launch_bounds__(NT) step_kernel(
    int t, const int* __restrict__ inputs, const float* __restrict__ W_emb,
    const float* __restrict__ b_ih, const float* __restrict__ b_hh,
    const unsigned short* __restrict__ wg_bf,
    const unsigned short* __restrict__ wencT_bf,
    const unsigned short* __restrict__ wdecT_bf,
    const float* __restrict__ h_e, const float* __restrict__ W_u,
    const float* __restrict__ b_u,
    float* h_g, float* c_g, float* E_hist, float* hd_hist,
    float* feat_f32, unsigned short* feat_bf, float* A_e,
    float* p_switch, float* row_sum)
{
  // LDS pool, manually carved; region A is reused: bf16 [x|h] -> f32 q -> f32 qd
  __shared__ __align__(16) unsigned char smem[16384 + 8192 + 25600 + 1536 + 64 + 64];
  unsigned short* s_xh = (unsigned short*)smem;                 // [16][512] bf16
  float*          s_q  = (float*)smem;                          // [16][256] f32 (alias)
  unsigned short* s_hb = (unsigned short*)(smem + 16384);       // [16][256] bf16
  float*          s_E  = (float*)(smem + 16384 + 8192);         // [16][400] f32
  float*          s_Ad = (float*)(smem + 16384 + 8192 + 25600); // [16][24]
  float*          s_rsum = (float*)(smem + 16384 + 8192 + 25600 + 1536);
  float*          s_dsum = s_rsum + 16;

  const int tid  = threadIdx.x;
  const int wave = tid >> 5;
  const int lane = tid & 31;
  const int hi   = lane >> 4;
  const int r16  = lane & 15;

  // ---- phase 0: stage A = [emb_t | h_{t-1}] as bf16 in LDS
  for (int i = tid; i < BSZ * KXH; i += NT) {
    int b = i >> 9, k = i & 511;
    float v;
    if (k < DIM) {
      int tok = inputs[t * BSZ + b];
      v = W_emb[(size_t)tok * DIM + k];
    } else {
      v = h_g[b * DIM + (k - DIM)];
    }
    s_xh[b * KXH + k] = f2bf(v);
  }
  __syncthreads();

  // ---- phase 1: LSTM gates via WMMA. Wave w owns d-slice [w*16, w*16+16);
  // it computes all four gate tiles (i,f,g,o) so the cell math stays in-register.
  {
    const int d0 = wave * 16;
    const unsigned short* a_base = s_xh + r16 * KXH + hi * 8;
    const unsigned short* bpi = wg_bf + (size_t)(0 * DIM + d0 + r16) * KXH + hi * 8;
    const unsigned short* bpf = wg_bf + (size_t)(1 * DIM + d0 + r16) * KXH + hi * 8;
    const unsigned short* bpg = wg_bf + (size_t)(2 * DIM + d0 + r16) * KXH + hi * 8;
    const unsigned short* bpo = wg_bf + (size_t)(3 * DIM + d0 + r16) * KXH + hi * 8;
    v8f ai = {}, af = {}, ag = {}, ao = {};
#pragma unroll
    for (int kb = 0; kb < KXH; kb += 32) {
      v16bf a = load_row16(a_base + kb);
      ai = __builtin_amdgcn_wmma_f32_16x16x32_bf16(false, a, false, load_row16(bpi + kb), (short)0, ai, false, false);
      af = __builtin_amdgcn_wmma_f32_16x16x32_bf16(false, a, false, load_row16(bpf + kb), (short)0, af, false, false);
      ag = __builtin_amdgcn_wmma_f32_16x16x32_bf16(false, a, false, load_row16(bpg + kb), (short)0, ag, false, false);
      ao = __builtin_amdgcn_wmma_f32_16x16x32_bf16(false, a, false, load_row16(bpo + kb), (short)0, ao, false, false);
    }
    const int d = d0 + r16;
    const float bi = b_ih[d]           + b_hh[d];
    const float bf = b_ih[DIM + d]     + b_hh[DIM + d];
    const float bg = b_ih[2 * DIM + d] + b_hh[2 * DIM + d];
    const float bo = b_ih[3 * DIM + d] + b_hh[3 * DIM + d];
#pragma unroll
    for (int r = 0; r < 8; ++r) {
      int m = r + hi * 8;                  // batch row
      int idx = m * DIM + d;
      float c_old = c_g[idx];
      float cnew = sigmoidf_fast(af[r] + bf) * c_old +
                   sigmoidf_fast(ai[r] + bi) * tanhf(ag[r] + bg);
      float hnew = sigmoidf_fast(ao[r] + bo) * tanhf(cnew);
      c_g[idx] = cnew;
      h_g[idx] = hnew;
      hd_hist[t * (BSZ * DIM) + idx] = hnew;
      feat_f32[m * FEAT + d] = hnew;
      feat_bf[m * FEAT + d] = f2bf(hnew);
      s_hb[m * DIM + d] = f2bf(hnew);
    }
  }
  __threadfence_block();
  __syncthreads();

  // ---- phase 2: q = h_t @ W_attn_enc via WMMA (one 16-wide j-tile per wave)
  {
    const int j0 = wave * 16;
    const unsigned short* a_base = s_hb + r16 * DIM + hi * 8;
    const unsigned short* bp = wencT_bf + (size_t)(j0 + r16) * DIM + hi * 8;
    v8f q = {};
#pragma unroll
    for (int kb = 0; kb < DIM; kb += 32) {
      q = __builtin_amdgcn_wmma_f32_16x16x32_bf16(false, load_row16(a_base + kb),
                                                  false, load_row16(bp + kb),
                                                  (short)0, q, false, false);
    }
#pragma unroll
    for (int r = 0; r < 8; ++r) s_q[(r + hi * 8) * DIM + j0 + r16] = q[r];
  }
  __syncthreads();

  // ---- phase 3: encoder attention energies (renormalized-history scheme)
  for (int i = tid; i < BSZ * SRCL; i += NT) {
    int b = i / SRCL, s = i % SRCL;
    const float* he = h_e + ((size_t)s * BSZ + b) * DIM;
    float dot = 0.f;
#pragma unroll 4
    for (int k = 0; k < DIM; ++k) dot += s_q[b * DIM + k] * he[k];
    float ex = __expf(dot);
    float Eh = E_hist[i];
    float EE = (t == 0) ? ex : (ex / Eh);
    E_hist[i] = (t == 0) ? ex : (Eh + ex);
    s_E[b * SRCL + s] = EE;
  }
  __syncthreads();

  // ---- phase 4: per-row sums of s_E (one wave per batch row)
  {
    float part = 0.f;
    for (int s = lane; s < SRCL; s += 32) part += s_E[wave * SRCL + s];
#pragma unroll
    for (int off = 1; off < 32; off <<= 1) part += __shfl_xor(part, off, 32);
    if (lane == 0) s_rsum[wave] = part;
  }
  __syncthreads();

  // ---- phase 5: A_e + context c_e -> feat[256:512]
  for (int i = tid; i < BSZ * SRCL; i += NT) {
    int b = i / SRCL, s = i % SRCL;
    float a = s_E[b * SRCL + s] / s_rsum[b];
    s_E[b * SRCL + s] = a;
    A_e[i] = a;
  }
  __syncthreads();
  for (int i = tid; i < BSZ * DIM; i += NT) {
    int b = i >> 8, d = i & 255;
    float acc = 0.f;
    for (int s = 0; s < SRCL; ++s)
      acc += s_E[b * SRCL + s] * h_e[((size_t)s * BSZ + b) * DIM + d];
    feat_f32[b * FEAT + DIM + d] = acc;
    feat_bf[b * FEAT + DIM + d] = f2bf(acc);
  }
  __syncthreads();

  // ---- phase 6: qd = h_t @ W_attn_dec via WMMA (overwrites s_q region)
  {
    const int j0 = wave * 16;
    const unsigned short* a_base = s_hb + r16 * DIM + hi * 8;
    const unsigned short* bp = wdecT_bf + (size_t)(j0 + r16) * DIM + hi * 8;
    v8f q = {};
#pragma unroll
    for (int kb = 0; kb < DIM; kb += 32) {
      q = __builtin_amdgcn_wmma_f32_16x16x32_bf16(false, load_row16(a_base + kb),
                                                  false, load_row16(bp + kb),
                                                  (short)0, q, false, false);
    }
#pragma unroll
    for (int r = 0; r < 8; ++r) s_q[(r + hi * 8) * DIM + j0 + r16] = q[r];
  }
  __syncthreads();

  // ---- phase 7: decoder self-attention over hd_hist[0..t]
  if (tid < BSZ * TGT) {
    int b = tid / TGT, tt = tid % TGT;
    float e = 0.f;
    if (tt <= t) {
      const float* hd = hd_hist + ((size_t)tt * BSZ + b) * DIM;
      float dot = 0.f;
#pragma unroll 4
      for (int k = 0; k < DIM; ++k) dot += s_q[b * DIM + k] * hd[k];
      e = __expf(dot);
    }
    s_Ad[b * TGT + tt] = e;
  }
  __syncthreads();
  if (tid < BSZ) {
    float s = 0.f;
    for (int tt = 0; tt <= t; ++tt) s += s_Ad[tid * TGT + tt];
    s_dsum[tid] = s;
  }
  __syncthreads();
  for (int i = tid; i < BSZ * DIM; i += NT) {
    int b = i >> 8, d = i & 255;
    float cdv = 0.f;
    if (t != 0) {
      float acc = 0.f;
      for (int tt = 0; tt <= t; ++tt)
        acc += s_Ad[b * TGT + tt] * hd_hist[((size_t)tt * BSZ + b) * DIM + d];
      cdv = acc / s_dsum[b];
    }
    feat_f32[b * FEAT + 2 * DIM + d] = cdv;
    feat_bf[b * FEAT + 2 * DIM + d] = f2bf(cdv);
  }
  __threadfence_block();
  __syncthreads();

  // ---- phase 8: p_switch = sigmoid(feat . w_u + b_u); zero row sums for logits pass
  {
    float part = 0.f;
    for (int k = lane; k < FEAT; k += 32) part += feat_f32[wave * FEAT + k] * W_u[k];
#pragma unroll
    for (int off = 1; off < 32; off <<= 1) part += __shfl_xor(part, off, 32);
    if (lane == 0) p_switch[wave] = sigmoidf_fast(part + b_u[0]);
    if (lane == 1) row_sum[wave] = 0.f;
  }
}

// ---------------- per-step big GEMM: exp(feat @ W_out^T + b) into d_out[t], + row sums
__global__ void logits_exp_kernel(const unsigned short* __restrict__ wout_bf,
                                  const unsigned short* __restrict__ feat_bf,
                                  const float* __restrict__ b_out,
                                  const int* __restrict__ pad_ptr,
                                  float* __restrict__ out_t,
                                  float* __restrict__ row_sum) {
  int wave = threadIdx.x >> 5;
  int lane = threadIdx.x & 31;
  int v0   = (blockIdx.x * 4 + wave) * 16;   // vocab tile base (2000 tiles)
  int hi   = lane >> 4;
  int r16  = lane & 15;
  const unsigned short* ap = feat_bf + r16 * FEAT + hi * 8;
  const unsigned short* bp = wout_bf + (size_t)(v0 + r16) * FEAT + hi * 8;
  __builtin_prefetch(bp, 0, 1);              // global_prefetch_b8 on the W_out stream
  v8f acc = {};
#pragma unroll
  for (int kb = 0; kb < FEAT; kb += 32) {
    v16bf a = load_row16(ap + kb);
    v16bf b = load_row16(bp + kb);
    acc = __builtin_amdgcn_wmma_f32_16x16x32_bf16(false, a, false, b,
                                                  (short)0, acc, false, false);
  }
  int col  = v0 + r16;
  int pad  = *pad_ptr;
  float bias = b_out[col];
#pragma unroll
  for (int r = 0; r < 8; ++r) {
    int m = r + hi * 8;                      // batch row of this accumulator
    float e = (col == pad) ? 0.0f : __expf(acc[r] + bias);
    out_t[(size_t)m * VOC + col] = e;
    float s = e;
#pragma unroll
    for (int off = 1; off < 16; off <<= 1) s += __shfl_xor(s, off, 32);
    if (r16 == 0) atomicAdd(&row_sum[m], s);
  }
}

// ---------------- normalize: scores = softmax * p_switch
__global__ void finalize_kernel(float* __restrict__ out_t,
                                const float* __restrict__ row_sum,
                                const float* __restrict__ p_switch) {
  int idx = blockIdx.x * blockDim.x + threadIdx.x;
  if (idx < BSZ * VOC) {
    int b = idx / VOC;
    out_t[idx] = out_t[idx] / row_sum[b] * p_switch[b];
  }
}

// ---------------- copy-distribution scatter-add
__global__ void copy_scatter_kernel(float* __restrict__ out_t,
                                    const int* __restrict__ src,
                                    const float* __restrict__ A_e,
                                    const float* __restrict__ p_switch) {
  int idx = blockIdx.x * blockDim.x + threadIdx.x;
  if (idx < BSZ * SRCL) {
    int b = idx / SRCL, s = idx % SRCL;
    int v = src[s * BSZ + b];                // src_ids = src.T
    atomicAdd(&out_t[(size_t)b * VOC + v], A_e[b * SRCL + s] * p_switch[b]);
  }
}

extern "C" void kernel_launch(void* const* d_in, const int* in_sizes, int n_in,
                              void* d_out, int out_size, void* d_ws, size_t ws_size,
                              hipStream_t stream) {
  const int*   inputs     = (const int*)d_in[0];
  const int*   src        = (const int*)d_in[1];
  const float* h_e        = (const float*)d_in[2];
  const float* h0         = (const float*)d_in[3];
  const float* c0         = (const float*)d_in[4];
  const float* W_emb      = (const float*)d_in[5];
  const float* W_ih       = (const float*)d_in[6];
  const float* b_ih       = (const float*)d_in[7];
  const float* W_hh       = (const float*)d_in[8];
  const float* b_hh       = (const float*)d_in[9];
  const float* W_attn_enc = (const float*)d_in[10];
  const float* W_attn_dec = (const float*)d_in[11];
  const float* W_proj     = (const float*)d_in[12];
  const float* W_u        = (const float*)d_in[13];
  const float* b_u        = (const float*)d_in[14];
  const float* b_out      = (const float*)d_in[15];
  const int*   pad_ptr    = (const int*)d_in[16];
  float* out = (float*)d_out;

  unsigned char* wsb = (unsigned char*)d_ws;
  size_t off = 0;
  auto carve = [&](size_t bytes) -> void* {
    void* p = wsb + off;
    off = (off + bytes + 255) & ~(size_t)255;
    return p;
  };
  unsigned short* emb_bf   = (unsigned short*)carve((size_t)VOC * DIM * 2);
  unsigned short* projT_bf = (unsigned short*)carve((size_t)FEAT * DIM * 2);
  unsigned short* wout_bf  = (unsigned short*)carve((size_t)VOC * FEAT * 2);
  unsigned short* wg_bf    = (unsigned short*)carve((size_t)GDIM * KXH * 2);
  unsigned short* wencT_bf = (unsigned short*)carve((size_t)DIM * DIM * 2);
  unsigned short* wdecT_bf = (unsigned short*)carve((size_t)DIM * DIM * 2);
  float* h_g      = (float*)carve((size_t)BSZ * DIM * 4);
  float* c_g      = (float*)carve((size_t)BSZ * DIM * 4);
  float* E_hist   = (float*)carve((size_t)BSZ * SRCL * 4);
  float* hd_hist  = (float*)carve((size_t)TGT * BSZ * DIM * 4);
  float* feat_f32 = (float*)carve((size_t)BSZ * FEAT * 4);
  unsigned short* feat_bf = (unsigned short*)carve((size_t)BSZ * FEAT * 2);
  float* A_e      = (float*)carve((size_t)BSZ * SRCL * 4);
  float* p_sw     = (float*)carve((size_t)BSZ * 4);
  float* row_sum  = (float*)carve((size_t)BSZ * 4);

  // one-time precompute
  const int n_conv = VOC * DIM + FEAT * DIM + GDIM * KXH + 2 * DIM * DIM;
  convert_kernel<<<(n_conv + 255) / 256, 256, 0, stream>>>(
      W_emb, W_proj, W_ih, W_hh, W_attn_enc, W_attn_dec,
      emb_bf, projT_bf, wg_bf, wencT_bf, wdecT_bf);
  wout_gemm_kernel<<<(VOC / 16) * (FEAT / 16) / 4, 128, 0, stream>>>(
      emb_bf, projT_bf, wout_bf);
  init_kernel<<<(TGT * BSZ * DIM + 255) / 256, 256, 0, stream>>>(
      h0, c0, h_g, c_g, E_hist, hd_hist);

  // 24 sequential decode steps
  for (int t = 0; t < TGT; ++t) {
    float* out_t = out + (size_t)t * BSZ * VOC;
    step_kernel<<<1, NT, 0, stream>>>(
        t, inputs, W_emb, b_ih, b_hh, wg_bf, wencT_bf, wdecT_bf,
        h_e, W_u, b_u, h_g, c_g, E_hist, hd_hist,
        feat_f32, feat_bf, A_e, p_sw, row_sum);
    logits_exp_kernel<<<(VOC / 16) / 4, 128, 0, stream>>>(
        wout_bf, feat_bf, b_out, pad_ptr, out_t, row_sum);
    finalize_kernel<<<(BSZ * VOC + 255) / 256, 256, 0, stream>>>(
        out_t, row_sum, p_sw);
    copy_scatter_kernel<<<(BSZ * SRCL + 255) / 256, 256, 0, stream>>>(
        out_t, src, A_e, p_sw);
  }
}
